// Seq2SeqWithAttention_56968446214262
// MI455X (gfx1250) — matrix operand order repaired and checked by reading, not compiled, not run on hardware
//
#include <hip/hip_runtime.h>
#include <hip/hip_bf16.h>
#include <stdint.h>

// ---------------------------------------------------------------------------
// Seq2Seq greedy decode with attention for MI455X (gfx1250, wave32, WMMA).
// Strategy:
//  * bf16 WMMA (v_wmma_f32_16x16x32_bf16) for every GEMM, f32 accumulate.
//  * One persistent decode kernel (32 blocks x 256 thr) with an atomic grid
//    barrier; each LSTM cell computes all 4 gates per (batch,d) tile in-wave.
//  * Weights pre-packed into per-lane-contiguous WMMA B fragments once.
//  * Tied-embedding logits + packed-u64 argmax (first-index tie-break).
// ---------------------------------------------------------------------------

typedef __attribute__((ext_vector_type(16))) __bf16   v16bf;
typedef __attribute__((ext_vector_type(8)))  float    v8f;
typedef __attribute__((ext_vector_type(4)))  unsigned v4u;

struct Pair32 { v4u lo; v4u hi; };

#define TQ   32
#define LQ   2
#define NBLK 32
#define BD   (32 * 512)   // elements in one [B][D] buffer

__device__ inline unsigned short f2bf_u16(float f) {
  unsigned u = __float_as_uint(f);
  unsigned r = (u + 0x7FFFu + ((u >> 16) & 1u)) >> 16;   // round-nearest-even
  return (unsigned short)r;
}
__device__ inline float bf2f(unsigned short h) { return __uint_as_float(((unsigned)h) << 16); }
__device__ inline float sigm(float x) { return 1.0f / (1.0f + __expf(-x)); }

// A fragment (16x32 bf16, M x K): layout — lane<16 holds K{0..15},
// lane>=16 holds K{16..31}; row-major bf16 source -> two b128 loads per lane.
__device__ inline v16bf load_a_frag(const unsigned short* base, int rowStrideWords,
                                    int mt, int kstep, int lane) {
  const int r = lane & 15, hf = lane >> 4;
  const unsigned* rp = reinterpret_cast<const unsigned*>(base)
                     + (size_t)(mt * 16 + r) * rowStrideWords + kstep * 16 + hf * 8;
  Pair32 pr;
  pr.lo = *reinterpret_cast<const v4u*>(rp);
  pr.hi = *reinterpret_cast<const v4u*>(rp + 4);
  return __builtin_bit_cast(v16bf, pr);
}
// B fragment: pre-packed per-lane-contiguous (8 dwords/lane = 2 x b128).
__device__ inline v16bf load_b_frag(const unsigned* frag, int lane) {
  const unsigned* rp = frag + lane * 8;
  Pair32 pr;
  pr.lo = *reinterpret_cast<const v4u*>(rp);
  pr.hi = *reinterpret_cast<const v4u*>(rp + 4);
  return __builtin_bit_cast(v16bf, pr);
}

__device__ inline v8f wmma_bf16(v16bf a, v16bf b, v8f c) {
  return __builtin_amdgcn_wmma_f32_16x16x32_bf16(false, a, false, b, (short)0, c, false, false);
}

// Monotonic-counter grid barrier (all NBLK blocks co-resident).
__device__ inline void grid_sync(unsigned* cnt, unsigned& calls) {
  __syncthreads();
  if (threadIdx.x == 0) {
    __threadfence();
    ++calls;
    atomicAdd(cnt, 1u);
    const unsigned target = (unsigned)NBLK * calls;
    while (atomicAdd(cnt, 0u) < target) __builtin_amdgcn_s_sleep(1);
    __threadfence();
  }
  __syncthreads();
}

__device__ inline float blk_max(float v, float* red) {
#pragma unroll
  for (int m = 16; m >= 1; m >>= 1) v = fmaxf(v, __shfl_xor(v, m, 32));
  if ((threadIdx.x & 31) == 0) red[threadIdx.x >> 5] = v;
  __syncthreads();
  float r = red[0];
#pragma unroll
  for (int i = 1; i < 8; ++i) r = fmaxf(r, red[i]);
  __syncthreads();
  return r;
}
__device__ inline float blk_sum(float v, float* red) {
#pragma unroll
  for (int m = 16; m >= 1; m >>= 1) v += __shfl_xor(v, m, 32);
  if ((threadIdx.x & 31) == 0) red[threadIdx.x >> 5] = v;
  __syncthreads();
  float r = 0.f;
#pragma unroll
  for (int i = 0; i < 8; ++i) r += red[i];
  __syncthreads();
  return r;
}

__device__ inline unsigned long long shfl_xor_u64_16(unsigned long long x, int m) {
  unsigned lo = (unsigned)x, hi = (unsigned)(x >> 32);
  lo = __shfl_xor(lo, m, 16);
  hi = __shfl_xor(hi, m, 16);
  return ((unsigned long long)hi << 32) | lo;
}

// --------------------------- weight packing --------------------------------
// B layout used here (contiguous k, same order as load_a_frag): lane<16 -> K 0..15
// (dword v packs K=2v,2v+1), lane>=16 -> K 16..31; N = lane&15.

__global__ void pack_lstm_kernel(const float* __restrict__ Wih,
                                 const float* __restrict__ Whh,
                                 unsigned* __restrict__ out) {
  const size_t idx = (size_t)blockIdx.x * 256 + threadIdx.x;  // 2,097,152
  const unsigned v  = idx & 7u;
  const unsigned ln = (idx >> 3) & 31u;
  const unsigned ks = (idx >> 8) & 31u;
  const unsigned nt = (idx >> 13) & 127u;
  const unsigned l  = (idx >> 20) & 1u;
  const int n  = nt * 16 + (ln & 15);
  const int k0 = ks * 32 + ((ln >> 4) ? 16 : 0) + 2 * (int)v;   // Kcat = [x(512) | h(512)]
  float a0, a1;
  if (k0 < 512) {
    const float* row = Wih + ((size_t)l * 2048 + n) * 512;
    a0 = row[k0]; a1 = row[k0 + 1];
  } else {
    const float* row = Whh + ((size_t)l * 2048 + n) * 512;
    a0 = row[k0 - 512]; a1 = row[k0 - 511];
  }
  out[idx] = (unsigned)f2bf_u16(a0) | ((unsigned)f2bf_u16(a1) << 16);
}

__global__ void pack_bT_kernel(const float* __restrict__ W, int K, int NT,
                               unsigned* __restrict__ out) {
  const size_t idx = (size_t)blockIdx.x * 256 + threadIdx.x;
  const unsigned v  = idx & 7u;
  const unsigned ln = (idx >> 3) & 31u;
  const size_t f  = idx >> 8;
  const int KS = K >> 5;
  const int ks = (int)(f % (size_t)KS);
  const int nt = (int)(f / (size_t)KS);
  if (nt >= NT) return;
  const int n  = nt * 16 + (ln & 15);
  const int k0 = ks * 32 + ((ln >> 4) ? 16 : 0) + 2 * (int)v;
  const float* row = W + (size_t)n * K;
  out[idx] = (unsigned)f2bf_u16(row[k0]) | ((unsigned)f2bf_u16(row[k0 + 1]) << 16);
}

__global__ void gather_srce_kernel(const int* __restrict__ src,
                                   const float* __restrict__ table,
                                   float* __restrict__ out) {
  const size_t idx = (size_t)blockIdx.x * 256 + threadIdx.x;  // 8,388,608
  const int d = (int)(idx & 511);
  const size_t bs = idx >> 9;
  const int tok = src[bs];
  out[idx] = table[(size_t)tok * 512 + d];
}

__global__ void init_state_kernel(const float* __restrict__ trg_emb,
                                  unsigned short* __restrict__ embbf,
                                  unsigned short* __restrict__ hbf,
                                  float* __restrict__ cst,
                                  unsigned long long* __restrict__ best,
                                  unsigned* __restrict__ cnt) {
  const int idx = blockIdx.x * 256 + threadIdx.x;  // 65536
  if (idx < 2 * 2 * 32 * 512) hbf[idx] = 0;
  if (idx < 2 * 32 * 512)     cst[idx] = 0.f;
  if (idx < 32)               best[idx] = 0ull;
  if (idx == 0)               *cnt = 0u;
  if (idx < 32 * 512) {
    const int b = idx >> 9, d = idx & 511;
    embbf[(size_t)b * 512 + d] = f2bf_u16(trg_emb[(size_t)1 * 512 + d]);  // SOS=1
  }
}

// ------------------------------ decode -------------------------------------

struct DecodeP {
  const float* trg_emb;     // [32000,512]
  const float* bih;         // [2,2048]
  const float* bhh;
  const float* bfc;         // [512]
  const float* bh2o;        // [512]
  const float* bout;        // [32000]
  const float* src_e;       // [32,512,512]
  const unsigned* wfragL;   // [2][128 nt][32 ks][256]
  const unsigned* wfcfrag;  // [32][16][256]
  const unsigned* wh2ofrag; // [32][32][256]
  const unsigned* embfrag;  // [2000][16][256]
  unsigned short* embbf;    // [T][32][512] bf16
  unsigned short* hbf;      // [2 layer][2 buf][32][512] bf16
  float* cst;               // [2][32][512]
  unsigned short* htop;     // [T][32][512] bf16 (filled at t==T-1)
  float* out_t;             // [32][512]
  unsigned short* xcat2;    // [32][1024] bf16  = [out_t | context]
  unsigned short* hid;      // [32][512] bf16
  unsigned long long* best; // [32]
  unsigned* cnt;
  float* out_tokens;        // d_out (tokens region, [32][32] as float)
};

__global__ __launch_bounds__(256) void decode_kernel(DecodeP p) {
  const int tid  = threadIdx.x;
  const int lane = tid & 31;
  const int wid  = blockIdx.x * 8 + (tid >> 5);   // 0..255
  const int hf   = lane >> 4;
  const int col  = lane & 15;
  __shared__ float sq[512];
  __shared__ float sw[512];
  __shared__ float sred[8];
  int cur = 0;                 // h ping-pong buffer holding current state
  unsigned calls = 0;

  for (int t = 0; t < TQ; ++t) {
    // ---------------- inner LSTM scan over active positions ----------------
    for (int pos = 0; pos <= t; ++pos) {
      const int nw = cur ^ 1;
      for (int l = 0; l < LQ; ++l) {
        if (wid < 64) {
          const int mt = wid & 1;      // batch tile 0..1
          const int dt = wid >> 1;     // hidden-unit tile 0..31
          const unsigned short* xsrc = (l == 0) ? (p.embbf + (size_t)pos * BD)
                                                : (p.hbf + (size_t)(0 * 2 + nw) * BD);
          const unsigned short* hsrc = p.hbf + (size_t)(l * 2 + cur) * BD;
          v8f a0 = {}, a1 = {}, a2 = {}, a3 = {};
          const unsigned* wb = p.wfragL + ((size_t)(l * 128 + dt) * 32) * 256;
          for (int ks = 0; ks < 32; ++ks) {
            v16bf af = (ks < 16) ? load_a_frag(xsrc, 256, mt, ks, lane)
                                 : load_a_frag(hsrc, 256, mt, ks - 16, lane);
            const unsigned* wk = wb + (size_t)ks * 256;
            v16bf b0 = load_b_frag(wk, lane);                       // gate i (nt=dt)
            v16bf b1 = load_b_frag(wk + 1u * 32u * 32u * 256u, lane); // gate f
            v16bf b2 = load_b_frag(wk + 2u * 32u * 32u * 256u, lane); // gate g
            v16bf b3 = load_b_frag(wk + 3u * 32u * 32u * 256u, lane); // gate o
            a0 = wmma_bf16(af, b0, a0);
            a1 = wmma_bf16(af, b1, a1);
            a2 = wmma_bf16(af, b2, a2);
            a3 = wmma_bf16(af, b3, a3);
          }
          const int dn = dt * 16 + col;
          const float biI = p.bih[l * 2048 + dn]        + p.bhh[l * 2048 + dn];
          const float biF = p.bih[l * 2048 + 512 + dn]  + p.bhh[l * 2048 + 512 + dn];
          const float biG = p.bih[l * 2048 + 1024 + dn] + p.bhh[l * 2048 + 1024 + dn];
          const float biO = p.bih[l * 2048 + 1536 + dn] + p.bhh[l * 2048 + 1536 + dn];
          unsigned short* hdst = p.hbf + (size_t)(l * 2 + nw) * BD;
          float* crow = p.cst + (size_t)l * BD;
#pragma unroll
          for (int r = 0; r < 8; ++r) {
            const int brow = mt * 16 + hf * 8 + r;   // documented C/D layout
            const float iv = sigm(a0[r] + biI);
            const float fv = sigm(a1[r] + biF);
            const float gv = tanhf(a2[r] + biG);
            const float ov = sigm(a3[r] + biO);
            float* cp = crow + (size_t)brow * 512 + dn;
            const float c2 = fv * (*cp) + iv * gv;
            const float h2 = ov * tanhf(c2);
            *cp = c2;                                   // c private to this wave
            hdst[(size_t)brow * 512 + dn] = f2bf_u16(h2);
            if (l == 1 && t == TQ - 1)
              p.htop[((size_t)pos * 32 + brow) * 512 + dn] = f2bf_u16(h2);
          }
        }
        grid_sync(p.cnt, calls);
      }
      cur = nw;
    }
    // ---------------- out_t = h_top @ Wfc^T + bfc --------------------------
    if (wid < 64) {
      const int mt = wid & 1;
      const int nt = wid >> 1;
      v8f acc = {};
      const unsigned short* A = p.hbf + (size_t)(1 * 2 + cur) * BD;
      for (int ks = 0; ks < 16; ++ks) {
        v16bf af = load_a_frag(A, 256, mt, ks, lane);
        v16bf bf_ = load_b_frag(p.wfcfrag + ((size_t)nt * 16 + ks) * 256, lane);
        acc = wmma_bf16(af, bf_, acc);
      }
      const int n = nt * 16 + col;
      const float bias = p.bfc[n];
#pragma unroll
      for (int r = 0; r < 8; ++r) {
        const int brow = mt * 16 + hf * 8 + r;
        const float v = acc[r] + bias;
        p.out_t[(size_t)brow * 512 + n] = v;
        p.xcat2[(size_t)brow * 1024 + n] = f2bf_u16(v);
      }
    }
    grid_sync(p.cnt, calls);
    // ---------------- attention + context (block b = blockIdx.x) -----------
    {
      const int b = blockIdx.x;                 // gridDim.x == 32 == B
      if (tid == 0) p.best[b] = 0ull;           // reset before this step's argmax
      for (int d0 = tid; d0 < 512; d0 += 256) sq[d0] = p.out_t[(size_t)b * 512 + d0];
      __syncthreads();
      float localmax = -3.0e38f;
#pragma unroll
      for (int s0 = 0; s0 < 2; ++s0) {
        const int s = tid * 2 + s0;
        const float* e = p.src_e + ((size_t)b * 512 + s) * 512;
        float a = 0.f;
        for (int d0 = 0; d0 < 512; d0 += 4)
          a += sq[d0] * e[d0] + sq[d0 + 1] * e[d0 + 1] + sq[d0 + 2] * e[d0 + 2] + sq[d0 + 3] * e[d0 + 3];
        a *= 0.04419417382415922f;  // 1/sqrt(512)
        sw[s] = a;
        localmax = fmaxf(localmax, a);
      }
      const float mx = blk_max(localmax, sred);
      float lsum = 0.f;
#pragma unroll
      for (int s0 = 0; s0 < 2; ++s0) {
        const int s = tid * 2 + s0;
        const float ev = __expf(sw[s] - mx);
        sw[s] = ev; lsum += ev;
      }
      const float tot = blk_sum(lsum, sred);
      const float inv = 1.0f / tot;
      __syncthreads();
      for (int d0 = tid; d0 < 512; d0 += 256) {
        const float* e = p.src_e + (size_t)b * 512 * 512 + d0;
        float a = 0.f;
        for (int s = 0; s < 512; ++s) a += sw[s] * e[(size_t)s * 512];
        p.xcat2[(size_t)b * 1024 + 512 + d0] = f2bf_u16(a * inv);
      }
    }
    grid_sync(p.cnt, calls);
    // ---------------- hid = relu([out_t|ctx] @ Wh2o^T + bh2o) --------------
    if (wid < 64) {
      const int mt = wid & 1;
      const int nt = wid >> 1;
      v8f acc = {};
      for (int ks = 0; ks < 32; ++ks) {
        v16bf af = load_a_frag(p.xcat2, 512, mt, ks, lane);
        v16bf bf_ = load_b_frag(p.wh2ofrag + ((size_t)nt * 32 + ks) * 256, lane);
        acc = wmma_bf16(af, bf_, acc);
      }
      const int n = nt * 16 + col;
      const float bias = p.bh2o[n];
#pragma unroll
      for (int r = 0; r < 8; ++r) {
        const int brow = mt * 16 + hf * 8 + r;
        float v = acc[r] + bias;
        v = v > 0.f ? v : 0.f;
        p.hid[(size_t)brow * 512 + n] = f2bf_u16(v);
      }
    }
    grid_sync(p.cnt, calls);
    // ---------------- logits (tied embedding) + argmax ---------------------
    {
      const int mt = wid & 1;
      unsigned long long best8[8];
#pragma unroll
      for (int r = 0; r < 8; ++r) best8[r] = 0ull;
      for (int nt = (wid >> 1); nt < 2000; nt += 128) {
        if (nt + 128 < 2000)
          __builtin_prefetch(p.embfrag + (size_t)(nt + 128) * 16 * 256, 0, 1);
        v8f acc = {};
        const unsigned* fb = p.embfrag + (size_t)nt * 16 * 256;
        for (int ks = 0; ks < 16; ++ks) {
          v16bf af = load_a_frag(p.hid, 256, mt, ks, lane);
          v16bf bf_ = load_b_frag(fb + (size_t)ks * 256, lane);
          acc = wmma_bf16(af, bf_, acc);
        }
        const int n = nt * 16 + col;
        const float bo = p.bout[n];
#pragma unroll
        for (int r = 0; r < 8; ++r) {
          const float v = acc[r] + bo;
          unsigned u = __float_as_uint(v);
          const unsigned key = (u & 0x80000000u) ? ~u : (u | 0x80000000u);  // order-preserving
          const unsigned long long pk = ((unsigned long long)key << 32) | (unsigned)(~(unsigned)n);
          if (pk > best8[r]) best8[r] = pk;     // ~n => first index wins ties
        }
      }
#pragma unroll
      for (int m = 1; m < 16; m <<= 1)
#pragma unroll
        for (int r = 0; r < 8; ++r) {
          const unsigned long long o = shfl_xor_u64_16(best8[r], m);
          if (o > best8[r]) best8[r] = o;
        }
      if (col == 0) {
        const int rowbase = mt * 16 + hf * 8;
#pragma unroll
        for (int r = 0; r < 8; ++r)
          atomicMax(&p.best[rowbase + r], best8[r]);
      }
    }
    grid_sync(p.cnt, calls);
    // ---------------- emit token, gather next-position embedding -----------
    {
      const int flat = blockIdx.x * 256 + tid;   // 8192 threads
      for (int e = flat; e < 32 * 512; e += NBLK * 256) {
        const int b = e >> 9, d = e & 511;
        const unsigned long long pk = p.best[b];
        const unsigned tok = ~(unsigned)(pk & 0xFFFFFFFFull);
        if (d == 0) p.out_tokens[b * TQ + t] = (float)tok;
        if (t + 1 < TQ)
          p.embbf[((size_t)(t + 1) * 32 + b) * 512 + d] =
              f2bf_u16(p.trg_emb[(size_t)tok * 512 + d]);
      }
    }
    grid_sync(p.cnt, calls);
  }
}

// --------------- final attention map: out_seqs[-1] vs src_e ----------------
__global__ __launch_bounds__(256) void attn_kernel(const unsigned short* __restrict__ htop,
                                                   const float* __restrict__ Wfc,
                                                   const float* __restrict__ bfc,
                                                   const float* __restrict__ src_e,
                                                   float* __restrict__ attn) {
  const int b  = blockIdx.x >> 5;   // 0..31
  const int ts = blockIdx.x & 31;   // position
  const int tid = threadIdx.x;
  __shared__ float hh[512];
  __shared__ float q[512];
  __shared__ float sc[512];
  __shared__ float red[8];
  for (int d = tid; d < 512; d += 256) hh[d] = bf2f(htop[((size_t)ts * 32 + b) * 512 + d]);
  __syncthreads();
  for (int d = tid; d < 512; d += 256) {
    const float* wr = Wfc + (size_t)d * 512;
    float a = 0.f;
    for (int k = 0; k < 512; k += 4)
      a += hh[k] * wr[k] + hh[k + 1] * wr[k + 1] + hh[k + 2] * wr[k + 2] + hh[k + 3] * wr[k + 3];
    q[d] = a + bfc[d];
  }
  __syncthreads();
  float localmax = -3.0e38f;
#pragma unroll
  for (int s0 = 0; s0 < 2; ++s0) {
    const int s = tid * 2 + s0;
    const float* e = src_e + ((size_t)b * 512 + s) * 512;
    float a = 0.f;
    for (int d = 0; d < 512; d += 4)
      a += q[d] * e[d] + q[d + 1] * e[d + 1] + q[d + 2] * e[d + 2] + q[d + 3] * e[d + 3];
    a *= 0.04419417382415922f;
    sc[s] = a; localmax = fmaxf(localmax, a);
  }
  const float mx = blk_max(localmax, red);
  float ls = 0.f;
#pragma unroll
  for (int s0 = 0; s0 < 2; ++s0) {
    const int s = tid * 2 + s0;
    const float ev = __expf(sc[s] - mx);
    sc[s] = ev; ls += ev;
  }
  const float tot = blk_sum(ls, red);
  const float inv = 1.0f / tot;
#pragma unroll
  for (int s0 = 0; s0 < 2; ++s0) {
    const int s = tid * 2 + s0;
    attn[(((size_t)b * 32) + ts) * 512 + s] = sc[s] * inv;
  }
}

// ------------------------------- host --------------------------------------
extern "C" void kernel_launch(void* const* d_in, const int* in_sizes, int n_in,
                              void* d_out, int out_size, void* d_ws, size_t ws_size,
                              hipStream_t stream) {
  (void)in_sizes; (void)n_in; (void)out_size; (void)ws_size;
  const int*   src     = (const int*)d_in[0];
  const float* src_emb = (const float*)d_in[1];
  const float* trg_emb = (const float*)d_in[2];
  const float* Wih     = (const float*)d_in[3];
  const float* Whh     = (const float*)d_in[4];
  const float* bih     = (const float*)d_in[5];
  const float* bhh     = (const float*)d_in[6];
  const float* Wfc     = (const float*)d_in[7];
  const float* bfc     = (const float*)d_in[8];
  const float* Wh2o    = (const float*)d_in[9];
  const float* bh2o    = (const float*)d_in[10];
  const float* bout    = (const float*)d_in[11];

  char* base = (char*)d_ws;
  auto carve = [&](size_t bytes) -> void* {
    void* p = (void*)base;
    base += (bytes + 255) & ~(size_t)255;
    return p;
  };
  float*              src_e  = (float*)carve((size_t)32 * 512 * 512 * 4);        // 32 MB
  unsigned*           wfragL = (unsigned*)carve((size_t)2 * 128 * 32 * 256 * 4); //  8 MB
  unsigned*           wfcf   = (unsigned*)carve((size_t)32 * 16 * 256 * 4);
  unsigned*           wh2of  = (unsigned*)carve((size_t)32 * 32 * 256 * 4);
  unsigned*           embf   = (unsigned*)carve((size_t)2000 * 16 * 256 * 4);    // 32 MB
  unsigned short*     embbf  = (unsigned short*)carve((size_t)32 * 32 * 512 * 2);
  unsigned short*     hbf    = (unsigned short*)carve((size_t)2 * 2 * 32 * 512 * 2);
  float*              cst    = (float*)carve((size_t)2 * 32 * 512 * 4);
  unsigned short*     htop   = (unsigned short*)carve((size_t)32 * 32 * 512 * 2);
  float*              out_t  = (float*)carve((size_t)32 * 512 * 4);
  unsigned short*     xcat2  = (unsigned short*)carve((size_t)32 * 1024 * 2);
  unsigned short*     hid    = (unsigned short*)carve((size_t)32 * 512 * 2);
  unsigned long long* best   = (unsigned long long*)carve(32 * 8);
  unsigned*           cnt    = (unsigned*)carve(256);

  pack_lstm_kernel<<<8192, 256, 0, stream>>>(Wih, Whh, wfragL);
  pack_bT_kernel<<<512, 256, 0, stream>>>(Wfc, 512, 32, wfcf);
  pack_bT_kernel<<<1024, 256, 0, stream>>>(Wh2o, 1024, 32, wh2of);
  pack_bT_kernel<<<32000, 256, 0, stream>>>(trg_emb, 512, 2000, embf);
  gather_srce_kernel<<<32768, 256, 0, stream>>>(src, src_emb, src_e);
  init_state_kernel<<<256, 256, 0, stream>>>(trg_emb, embbf, hbf, cst, best, cnt);

  DecodeP p;
  p.trg_emb  = trg_emb;
  p.bih      = bih;   p.bhh = bhh;
  p.bfc      = bfc;   p.bh2o = bh2o;  p.bout = bout;
  p.src_e    = src_e;
  p.wfragL   = wfragL; p.wfcfrag = wfcf; p.wh2ofrag = wh2of; p.embfrag = embf;
  p.embbf    = embbf;  p.hbf = hbf;  p.cst = cst;  p.htop = htop;
  p.out_t    = out_t;  p.xcat2 = xcat2; p.hid = hid;
  p.best     = best;   p.cnt = cnt;
  p.out_tokens = (float*)d_out;

  decode_kernel<<<NBLK, 256, 0, stream>>>(p);
  attn_kernel<<<32 * 32, 256, 0, stream>>>(htop, Wfc, bfc, src_e,
                                           (float*)d_out + 32 * 32);
}
